// standalone_self_attention_1D_48189533061166
// MI455X (gfx1250) — matrix operand order repaired and checked by reading, not compiled, not run on hardware
//
#include <hip/hip_runtime.h>
#include <hip/hip_bf16.h>
#include <math.h>

typedef __attribute__((ext_vector_type(16))) __bf16 v16bf;
typedef __attribute__((ext_vector_type(8)))  __bf16 bfx8;
typedef __attribute__((ext_vector_type(8)))  float  v8f;
typedef __attribute__((ext_vector_type(4)))  float  fx4;

constexpr int B     = 8;
constexpr int S     = 4096;             // power of two (q-row remap is shift/mask)
constexpr int F_IN  = 256;
constexpr int C     = 256;
constexpr int G     = 8;
constexpr int K     = 31;
constexpr int F_OUT = 88;
constexpr int NPADL = 96;               // F_OUT padded to 6x16 tiles
constexpr int PAD   = (K - 1) / 2;      // 15
constexpr int D     = C / G;            // 32 per-head dim
constexpr int SP    = S + 2 * PAD;      // 4126 padded frames
constexpr int MQ    = B * S;            // 32768 rows
constexpr int MP    = B * SP;           // 33008 rows (multiple of 16)
constexpr int MITER = 8;                // M-tiles per wave (B-frags reused)

// ---------------------------------------------------------------------------
// Prep 1: Wt[n][k] = (bf16)W[k][n], zero-padded in n up to Npad. Every WMMA
// B-fragment becomes one contiguous 32-byte chunk.
// ---------------------------------------------------------------------------
__global__ void transpose_w_bf16(const float* __restrict__ W,
                                 unsigned short* __restrict__ Wtu,
                                 int Kdim, int N, int Npad)
{
    __bf16* Wt = reinterpret_cast<__bf16*>(Wtu);
    int idx = blockIdx.x * blockDim.x + threadIdx.x;   // over Npad*Kdim
    if (idx >= Npad * Kdim) return;
    int n = idx / Kdim;
    int k = idx - n * Kdim;
    float v = (n < N) ? W[(long)k * N + n] : 0.f;
    Wt[idx] = (__bf16)v;
}

// ---------------------------------------------------------------------------
// Prep 2: Xp[B*SP, F_IN] bf16 = spec zero-padded by PAD frames per batch.
// Converts f32->bf16 exactly once; all three projection GEMMs then read pure
// bf16 with no masking/conversion in their inner loops.
// ---------------------------------------------------------------------------
__global__ void pad_spec_bf16(const float* __restrict__ spec,
                              unsigned short* __restrict__ Xpu)
{
    __bf16* Xp = reinterpret_cast<__bf16*>(Xpu);
    long idx = (long)(blockIdx.x * (long)blockDim.x + threadIdx.x) * 8;
    if (idx >= (long)MP * F_IN) return;
    long row = idx >> 8;                               // /F_IN
    int  col = (int)(idx & (F_IN - 1));
    int  b   = (int)(row / SP);
    int  s   = (int)(row - (long)b * SP) - PAD;

    bfx8 o;
    if ((unsigned)s < (unsigned)S) {
        const float* src = spec + ((long)(b * S + s) * F_IN + col);
        fx4 x0 = *(const fx4*)src;
        fx4 x1 = *(const fx4*)(src + 4);
        #pragma unroll
        for (int i = 0; i < 4; ++i) {
            o[i]     = (__bf16)x0[i];
            o[4 + i] = (__bf16)x1[i];
        }
    } else {
        #pragma unroll
        for (int i = 0; i < 8; ++i) o[i] = (__bf16)0.f;
    }
    *(bfx8*)(Xp + idx) = o;
}

// ---------------------------------------------------------------------------
// Kernel A: Y(bf16)[Mtiles*16, C] = Xp(bf16) @ Wt(bf16^T), pure-bf16 WMMA.
// One wave per 16-wide N column; B fragments (full K=256 column, 64 VGPRs)
// preloaded once and reused across MITER M-tiles. qmode remaps output row
// (b,s) -> padded row b*SP + s + PAD (shift/mask; S is a power of two).
// ---------------------------------------------------------------------------
__global__ void qkv_gemm_bf16(const unsigned short* __restrict__ Xpu,
                              const unsigned short* __restrict__ Wtu,
                              unsigned short* __restrict__ Yu,
                              int Mtiles, int qmode)
{
    const __bf16* Xp = reinterpret_cast<const __bf16*>(Xpu);
    const __bf16* Wt = reinterpret_cast<const __bf16*>(Wtu);
    __bf16*       Y  = reinterpret_cast<__bf16*>(Yu);

    const int lane   = threadIdx.x & 31;
    const int wave   = threadIdx.x >> 5;
    const int ntile  = blockIdx.x;
    const int mtile0 = (blockIdx.y * 4 + wave) * MITER;
    const int m  = lane & 15;
    const int hf = lane >> 4;
    const int n  = ntile * 16 + m;

    // B fragments: elements 0..15 = K (hf*16 + 0..15), contiguous in Wt.
    v16bf bfrag[8];
    {
        const __bf16* wrow = Wt + (long)n * F_IN;
        #pragma unroll
        for (int kt = 0; kt < 8; ++kt) {
            bfx8 lo = *(const bfx8*)(wrow + kt * 32 + hf * 16);
            bfx8 hi = *(const bfx8*)(wrow + kt * 32 + hf * 16 + 8);
            #pragma unroll
            for (int i = 0; i < 8; ++i) {
                bfrag[kt][i]     = lo[i];
                bfrag[kt][8 + i] = hi[i];
            }
        }
    }

    for (int it = 0; it < MITER; ++it) {
        const int mtile = mtile0 + it;
        if (mtile >= Mtiles) return;                 // wave-uniform
        const int r = mtile * 16 + m;

        int xrow;
        if (qmode) {
            int b = r >> 12;                         // r / S
            int s = r & (S - 1);
            xrow = b * SP + s + PAD;
        } else {
            xrow = r;
        }
        const __bf16* xr = Xp + (long)xrow * F_IN;
        // speculative prefetch of next M-tile's row (global_prefetch_b8)
        __builtin_prefetch(xr + 16 * F_IN, 0, 1);

        v8f acc = {};
        #pragma unroll
        for (int kt = 0; kt < 8; ++kt) {
            // A fragment: elems 0..15 = K (hf*16 + 0..15)
            bfx8 lo = *(const bfx8*)(xr + kt * 32 + hf * 16);
            bfx8 hi = *(const bfx8*)(xr + kt * 32 + hf * 16 + 8);
            v16bf a;
            #pragma unroll
            for (int i = 0; i < 8; ++i) {
                a[i]     = lo[i];
                a[8 + i] = hi[i];
            }
            acc = __builtin_amdgcn_wmma_f32_16x16x32_bf16(
                      false, a, false, bfrag[kt], (short)0, acc, false, false);
        }
        // C/D: element i -> row i + 8*hf, col = m
        #pragma unroll
        for (int i = 0; i < 8; ++i) {
            int crow = mtile * 16 + i + 8 * hf;
            Y[(long)crow * C + n] = (__bf16)acc[i];
        }
    }
}

// ---------------------------------------------------------------------------
// Kernel B: per-(b,s,g) windowed attention. e = q.k_win + q.rel (rel read
// contiguously per d-row), softmax(K=31), ctx = attn.v_win. q/k/v rows read
// as 16-byte bf16x8 vectors (64B-aligned head slices, all L2-resident).
// ---------------------------------------------------------------------------
__global__ void local_attn(const unsigned short* __restrict__ qu,
                           const unsigned short* __restrict__ ku,
                           const unsigned short* __restrict__ vu,
                           const float* __restrict__ rel,      // [C, K]
                           float* __restrict__ attn_out,       // [B*S*G, K]
                           unsigned short* __restrict__ ctxu)  // [B*S, C] bf16
{
    const __bf16* qb  = reinterpret_cast<const __bf16*>(qu);
    const __bf16* kb  = reinterpret_cast<const __bf16*>(ku);
    const __bf16* vb  = reinterpret_cast<const __bf16*>(vu);
    __bf16*       ctx = reinterpret_cast<__bf16*>(ctxu);

    int tid = blockIdx.x * blockDim.x + threadIdx.x;   // ((b*S+s)*G + g)
    if (tid >= B * S * G) return;
    int g  = tid & (G - 1);
    int bs = tid / G;
    int b  = bs >> 12;                                 // bs / S
    int s  = bs & (S - 1);

    float qv[D];
    {
        const bfx8* qp = (const bfx8*)(qb + (long)bs * C + g * D);
        #pragma unroll
        for (int c = 0; c < 4; ++c) {
            bfx8 t = qp[c];
            #pragma unroll
            for (int i = 0; i < 8; ++i) qv[c * 8 + i] = (float)t[i];
        }
    }

    // positional-bias term: e[k] = sum_d qv[d] * rel[g*D+d][k]  (contiguous rows)
    float e[K];
    #pragma unroll
    for (int k = 0; k < K; ++k) e[k] = 0.f;
    #pragma unroll
    for (int d = 0; d < D; ++d) {
        const float* rr = rel + (long)(g * D + d) * K;
        float qd = qv[d];
        #pragma unroll
        for (int k = 0; k < K; ++k) e[k] += qd * rr[k];
    }

    float emax = -3.0e38f;
    #pragma unroll
    for (int k = 0; k < K; ++k) {
        const bfx8* kp = (const bfx8*)(kb + ((long)(b * SP + s + k)) * C + g * D);
        float acc = e[k];
        #pragma unroll
        for (int c = 0; c < 4; ++c) {
            bfx8 t = kp[c];
            #pragma unroll
            for (int i = 0; i < 8; ++i) acc += qv[c * 8 + i] * (float)t[i];
        }
        e[k] = acc;
        emax = fmaxf(emax, acc);
    }
    float esum = 0.f;
    #pragma unroll
    for (int k = 0; k < K; ++k) { e[k] = __expf(e[k] - emax); esum += e[k]; }
    const float inv = 1.f / esum;
    float* ap = attn_out + (long)tid * K;
    #pragma unroll
    for (int k = 0; k < K; ++k) { e[k] *= inv; ap[k] = e[k]; }

    float o[D];
    #pragma unroll
    for (int d = 0; d < D; ++d) o[d] = 0.f;
    #pragma unroll
    for (int k = 0; k < K; ++k) {
        const bfx8* vp = (const bfx8*)(vb + ((long)(b * SP + s + k)) * C + g * D);
        float w = e[k];
        #pragma unroll
        for (int c = 0; c < 4; ++c) {
            bfx8 t = vp[c];
            #pragma unroll
            for (int i = 0; i < 8; ++i) o[c * 8 + i] += w * (float)t[i];
        }
    }
    __bf16* cp = ctx + (long)bs * C + g * D;
    #pragma unroll
    for (int d = 0; d < D; ++d) cp[d] = (__bf16)o[d];
}

// ---------------------------------------------------------------------------
// Kernel C: pred = sigmoid(ctx @ W_lin + b_lin). W_lin pre-transposed &
// zero-padded to [96][256] bf16; all loads unconditional; every WMMA runs
// with EXEC all-ones; only the final store is column-guarded.
// ---------------------------------------------------------------------------
__global__ void out_linear(const unsigned short* __restrict__ ctxu,
                           const unsigned short* __restrict__ Wtu,  // [96][256] bf16
                           const float* __restrict__ bias,          // [88]
                           float* __restrict__ pred)                // [MQ, 88]
{
    const __bf16* ctx = reinterpret_cast<const __bf16*>(ctxu);
    const __bf16* Wt  = reinterpret_cast<const __bf16*>(Wtu);

    const int lane   = threadIdx.x & 31;
    const int wave   = threadIdx.x >> 5;
    const int ntile  = blockIdx.x;                    // 0..5
    const int mtile0 = (blockIdx.y * 4 + wave) * MITER;
    const int m  = lane & 15;
    const int hf = lane >> 4;
    const int n  = ntile * 16 + m;                    // < 96 always

    v16bf bfrag[8];
    {
        const __bf16* wrow = Wt + (long)n * C;
        #pragma unroll
        for (int kt = 0; kt < 8; ++kt) {
            bfx8 lo = *(const bfx8*)(wrow + kt * 32 + hf * 16);
            bfx8 hi = *(const bfx8*)(wrow + kt * 32 + hf * 16 + 8);
            #pragma unroll
            for (int i = 0; i < 8; ++i) {
                bfrag[kt][i]     = lo[i];
                bfrag[kt][8 + i] = hi[i];
            }
        }
    }

    for (int it = 0; it < MITER; ++it) {
        const int mtile = mtile0 + it;               // MQ/16 = 2048, always valid
        const int row = mtile * 16 + m;
        const __bf16* crow = ctx + (long)row * C;
        __builtin_prefetch(crow + 16 * C, 0, 1);

        v8f acc = {};
        #pragma unroll
        for (int kt = 0; kt < 8; ++kt) {
            bfx8 lo = *(const bfx8*)(crow + kt * 32 + hf * 16);
            bfx8 hi = *(const bfx8*)(crow + kt * 32 + hf * 16 + 8);
            v16bf a;
            #pragma unroll
            for (int i = 0; i < 8; ++i) {
                a[i]     = lo[i];
                a[8 + i] = hi[i];
            }
            acc = __builtin_amdgcn_wmma_f32_16x16x32_bf16(
                      false, a, false, bfrag[kt], (short)0, acc, false, false);
        }
        if (n < F_OUT) {                             // divergent only at store
            const float bv = bias[n];
            #pragma unroll
            for (int i = 0; i < 8; ++i) {
                int r = mtile * 16 + i + 8 * hf;
                float x = acc[i] + bv;
                pred[(long)r * F_OUT + n] = 1.f / (1.f + __expf(-x));
            }
        }
    }
}

// ---------------------------------------------------------------------------
extern "C" void kernel_launch(void* const* d_in, const int* in_sizes, int n_in,
                              void* d_out, int out_size, void* d_ws, size_t ws_size,
                              hipStream_t stream)
{
    const float* spec  = (const float*)d_in[0];
    const float* W_q   = (const float*)d_in[1];
    const float* W_k   = (const float*)d_in[2];
    const float* W_v   = (const float*)d_in[3];
    const float* rel   = (const float*)d_in[4];
    const float* W_lin = (const float*)d_in[5];
    const float* b_lin = (const float*)d_in[6];

    float* pred = (float*)d_out;                       // [B,S,F_OUT]
    float* attn = pred + (size_t)MQ * F_OUT;           // [B,S,G,K]

    // workspace layout (all 16B-aligned chunks); Xp placed first so GEMM
    // prefetch overrun stays inside the allocation.
    char* p = (char*)d_ws;
    unsigned short* Xp  = (unsigned short*)p;  p += (size_t)MP * F_IN * 2;    // 16.9 MB
    unsigned short* WtQ = (unsigned short*)p;  p += (size_t)C * F_IN * 2;     // 128 KB
    unsigned short* WtK = (unsigned short*)p;  p += (size_t)C * F_IN * 2;
    unsigned short* WtV = (unsigned short*)p;  p += (size_t)C * F_IN * 2;
    unsigned short* WtL = (unsigned short*)p;  p += (size_t)NPADL * C * 2;    // 48 KB
    unsigned short* qb  = (unsigned short*)p;  p += (size_t)MQ * C * 2;       // 16 MB
    unsigned short* kb  = (unsigned short*)p;  p += (size_t)MP * C * 2;
    unsigned short* vb  = (unsigned short*)p;  p += (size_t)MP * C * 2;
    unsigned short* ctx = (unsigned short*)p;

    // prep (tiny)
    transpose_w_bf16<<<(C * F_IN + 255) / 256, 256, 0, stream>>>(W_q, WtQ, F_IN, C, C);
    transpose_w_bf16<<<(C * F_IN + 255) / 256, 256, 0, stream>>>(W_k, WtK, F_IN, C, C);
    transpose_w_bf16<<<(C * F_IN + 255) / 256, 256, 0, stream>>>(W_v, WtV, F_IN, C, C);
    transpose_w_bf16<<<(NPADL * C + 255) / 256, 256, 0, stream>>>(W_lin, WtL, C, F_OUT, NPADL);
    pad_spec_bf16<<<(int)(((long)MP * F_IN / 8 + 255) / 256), 256, 0, stream>>>(spec, Xp);

    const dim3 blk(128);   // 4 waves per block, 1 N-column block x MITER M-tiles each

    const int qjobs = (MQ / 16 + MITER - 1) / MITER;   // 256
    const int pjobs = (MP / 16 + MITER - 1) / MITER;   // 258
    qkv_gemm_bf16<<<dim3(C / 16, (qjobs + 3) / 4), blk, 0, stream>>>(
        Xp, WtQ, qb, MQ / 16, 1);
    qkv_gemm_bf16<<<dim3(C / 16, (pjobs + 3) / 4), blk, 0, stream>>>(
        Xp, WtK, kb, MP / 16, 0);
    qkv_gemm_bf16<<<dim3(C / 16, (pjobs + 3) / 4), blk, 0, stream>>>(
        Xp, WtV, vb, MP / 16, 0);

    local_attn<<<(B * S * G + 255) / 256, 256, 0, stream>>>(
        qb, kb, vb, rel, attn, ctx);

    const int ljobs = (MQ / 16) / MITER;               // 256
    out_linear<<<dim3(NPADL / 16, ljobs / 4), blk, 0, stream>>>(
        ctx, WtL, b_lin, pred);
}
